// DevignModel_19645180412194
// MI455X (gfx1250) — matrix operand order repaired and checked by reading, not compiled, not run on hardware
//
#include <hip/hip_runtime.h>
#include <hip/hip_bf16.h>

// ---------------------------------------------------------------------------
// Problem constants (from reference)
// ---------------------------------------------------------------------------
#define BB   64
#define NN   510
#define DIN  256
#define DD   256
#define TT   4
#define NSTEPS 6
#define EPG  (NN * 12)
#define ETOT (BB * EPG)          // 391680
#define BN   (BB * NN)           // 32640
#define CCH  (DIN + DD)          // 512

typedef __attribute__((ext_vector_type(16))) __bf16 v16bf;
typedef __attribute__((ext_vector_type(8)))  __bf16 v8bf;
typedef __attribute__((ext_vector_type(8)))  float  v8f;

// ---------------------------------------------------------------------------
// WMMA helpers (wave32, v_wmma_f32_16x16x32_bf16)
//  A 16x32: lane L: M=L&15 ; kb=(L&16)?16:0 ; subword g*8+j -> K = kb+g*8+j
//  B 32x16: lane L: N=L&15 ; kb=(L&16)?16:0 ; subword e    -> K = kb+e
//  C/D    : lane L: N=L&15 ; VGPR r -> M = r + ((L&16)?8:0)
// ---------------------------------------------------------------------------
__device__ inline v8f wmma_bf16(v16bf a, v16bf b, v8f c) {
  return __builtin_amdgcn_wmma_f32_16x16x32_bf16(false, a, false, b, (short)0, c,
                                                 false, false);
}

// A fragment from plain row-major bf16 [rows x ld]: two contiguous 16B loads/lane.
__device__ inline v16bf load_a_bf(const __bf16* __restrict__ X, int ld, int row0,
                                  int rowMax, int k0) {
  int lane = threadIdx.x & 31;
  int m = lane & 15;
  int kb = (lane & 16) ? 16 : 0;
  int row = row0 + m;
  if (row > rowMax) row = rowMax;
  const __bf16* p = X + (size_t)row * ld + k0 + kb;
  v8bf lo = *(const v8bf*)(p);        // K = kb .. kb+7       (g=0)
  v8bf hi = *(const v8bf*)(p + 8);    // K = kb+8 .. kb+15    (g=1)
  v16bf A;
#pragma unroll
  for (int j = 0; j < 8; ++j) { A[j] = lo[j]; A[8 + j] = hi[j]; }
  return A;
}

// B fragment from pre-packed tile blocks: one contiguous 32B load per lane.
__device__ inline v16bf load_b_packed(const __bf16* __restrict__ Wp, size_t blk) {
  int lane = threadIdx.x & 31;
  return ((const v16bf*)Wp)[blk * 32 + lane];
}

__device__ inline float sigmoidf_(float x) { return 1.f / (1.f + __expf(-x)); }

// ---------------------------------------------------------------------------
// Weight packer: W (fp32, layout [col][.] with rowStride/kStride, Ktap taps)
//   -> bf16 blocks [k][ct][kc][lane][16] matching the B-fragment layout.
// ---------------------------------------------------------------------------
__global__ void pack_b_kernel(const float* __restrict__ W, __bf16* __restrict__ out,
                              int otiles, int kchunks, int Ktap, int rowStride,
                              int kStride) {
  int tid = blockIdx.x * blockDim.x + threadIdx.x;
  int total = Ktap * otiles * kchunks * 32;
  if (tid >= total) return;
  int lane = tid & 31;
  int rest = tid >> 5;
  int kc = rest % kchunks; rest /= kchunks;
  int ct = rest % otiles;
  int k = rest / otiles;
  int n = lane & 15;
  int kbl = (lane & 16) ? 16 : 0;
  const float* src = W + (size_t)(ct * 16 + n) * rowStride +
                     (size_t)(kc * 32 + kbl) * kStride + k;
  __bf16* dst = out + ((size_t)((k * otiles + ct) * kchunks + kc) * 32 + lane) * 16;
#pragma unroll
  for (int e = 0; e < 16; ++e) dst[e] = (__bf16)src[(size_t)e * kStride];
}

// fp32 -> bf16 row-major shadow
__global__ void pack_a_kernel(const float* __restrict__ in, __bf16* __restrict__ out,
                              size_t total) {
  size_t i = (size_t)blockIdx.x * blockDim.x + threadIdx.x;
  if (i < total) out[i] = (__bf16)in[i];
}

// ---------------------------------------------------------------------------
// Generic conv1d (VALID) / GEMM via WMMA, bf16 A (row-major) + packed bf16 B.
//   Y[b,l,o] = act( bias[o] + sum_k sum_i X[b,l+k,i] * W[o,i,k] )
// Register-blocked: one wave computes a 16(l) x 64(o) tile (4 accumulators),
// sharing each A fragment across 4 WMMAs. Requires Cout % 64 == 0.
// Ktap==1 => plain GEMM. 8 waves/block.
// ---------------------------------------------------------------------------
__global__ __launch_bounds__(256) void conv1d_wmma_kernel(
    const __bf16* __restrict__ X, const __bf16* __restrict__ Wp,
    const float* __restrict__ bias, float* __restrict__ Y,
    int Bn, int Lin, int Cin, int Cout, int Ktap, int doRelu) {
  int Lout = Lin - Ktap + 1;
  int ltiles = (Lout + 15) >> 4;
  int otiles = Cout >> 4;   // 16-wide B tiles
  int ogroups = Cout >> 6;  // 64-wide output groups (4 tiles)
  int kchunks = Cin >> 5;
  int tile = blockIdx.x * (blockDim.x >> 5) + (threadIdx.x >> 5);
  if (tile >= Bn * ltiles * ogroups) return;
  int og = tile % ogroups;
  int tmp = tile / ogroups;
  int lt = tmp % ltiles;
  int b = tmp / ltiles;
  int l0 = lt << 4, o0 = og << 6;
  const __bf16* Xb = X + (size_t)b * Lin * Cin;

  int lane = threadIdx.x & 31;
  int n = lane & 15;
  v8f c0, c1, c2, c3;
  float b0 = bias[o0 + n], b1 = bias[o0 + 16 + n];
  float b2 = bias[o0 + 32 + n], b3 = bias[o0 + 48 + n];
#pragma unroll
  for (int r = 0; r < 8; ++r) { c0[r] = b0; c1[r] = b1; c2[r] = b2; c3[r] = b3; }

  for (int k = 0; k < Ktap; ++k) {
    size_t blk = ((size_t)k * otiles + (size_t)og * 4) * kchunks;
    for (int kc = 0; kc < kchunks; ++kc) {
      v16bf A = load_a_bf(Xb, Cin, l0 + k, Lin - 1, kc * 32);
      c0 = wmma_bf16(A, load_b_packed(Wp, blk + 0 * kchunks + kc), c0);
      c1 = wmma_bf16(A, load_b_packed(Wp, blk + 1 * kchunks + kc), c1);
      c2 = wmma_bf16(A, load_b_packed(Wp, blk + 2 * kchunks + kc), c2);
      c3 = wmma_bf16(A, load_b_packed(Wp, blk + 3 * kchunks + kc), c3);
    }
  }

  int mb = (lane & 16) ? 8 : 0;
#pragma unroll
  for (int r = 0; r < 8; ++r) {
    int l = l0 + mb + r;
    if (l < Lout) {
      float v0 = c0[r], v1 = c1[r], v2 = c2[r], v3 = c3[r];
      if (doRelu) {
        v0 = v0 > 0.f ? v0 : 0.f;
        v1 = v1 > 0.f ? v1 : 0.f;
        v2 = v2 > 0.f ? v2 : 0.f;
        v3 = v3 > 0.f ? v3 : 0.f;
      }
      float* yp = Y + ((size_t)b * Lout + l) * Cout + o0 + n;
      yp[0] = v0; yp[16] = v1; yp[32] = v2; yp[48] = v3;
    }
  }
}

// ---------------------------------------------------------------------------
// Fused GRU cell over [BN x 256] (bf16 A operands, packed bf16 weights):
//   r = sig(ir+hr); z = sig(iz+hz); n = tanh(in + r*hn); Hout = (1-z)*n + z*H
// A fragments shared across 3 gate WMMAs each.
// ---------------------------------------------------------------------------
__global__ __launch_bounds__(256) void gru_wmma_kernel(
    const __bf16* __restrict__ Abf, const __bf16* __restrict__ Hbf,
    const float* __restrict__ H, const __bf16* __restrict__ WihP,
    const __bf16* __restrict__ WhhP, const float* __restrict__ bih,
    const float* __restrict__ bhh, float* __restrict__ Hout, int M) {
  const int ntiles = DD >> 4;   // 16 col tiles
  const int kchunks = DD >> 5;  // 8
  int tile = blockIdx.x * (blockDim.x >> 5) + (threadIdx.x >> 5);
  if (tile >= (M >> 4) * ntiles) return;
  int ct = tile % ntiles, rt = tile / ntiles;
  int row0 = rt << 4, col0 = ct << 4;
  int lane = threadIdx.x & 31, n = lane & 15;

  v8f cir, ciz, cnn, dhr, dhz, dhn;
  float bir = bih[col0 + n], biz = bih[256 + col0 + n], bin = bih[512 + col0 + n];
  float bhr = bhh[col0 + n], bhz = bhh[256 + col0 + n], bhn = bhh[512 + col0 + n];
#pragma unroll
  for (int r = 0; r < 8; ++r) {
    cir[r] = bir; ciz[r] = biz; cnn[r] = bin;
    dhr[r] = bhr; dhz[r] = bhz; dhn[r] = bhn;
  }

  for (int kc = 0; kc < kchunks; ++kc) {
    v16bf Aa = load_a_bf(Abf, DD, row0, M - 1, kc * 32);
    v16bf Ah = load_a_bf(Hbf, DD, row0, M - 1, kc * 32);
    size_t br = (size_t)(ct + 0) * kchunks + kc;
    size_t bz = (size_t)(ct + 16) * kchunks + kc;
    size_t bn_ = (size_t)(ct + 32) * kchunks + kc;
    cir = wmma_bf16(Aa, load_b_packed(WihP, br), cir);
    ciz = wmma_bf16(Aa, load_b_packed(WihP, bz), ciz);
    cnn = wmma_bf16(Aa, load_b_packed(WihP, bn_), cnn);
    dhr = wmma_bf16(Ah, load_b_packed(WhhP, br), dhr);
    dhz = wmma_bf16(Ah, load_b_packed(WhhP, bz), dhz);
    dhn = wmma_bf16(Ah, load_b_packed(WhhP, bn_), dhn);
  }

  int mb = (lane & 16) ? 8 : 0;
#pragma unroll
  for (int r = 0; r < 8; ++r) {
    size_t idx = (size_t)(row0 + mb + r) * DD + col0 + n;
    float hv = H[idx];
    float rr = sigmoidf_(cir[r] + dhr[r]);
    float zz = sigmoidf_(ciz[r] + dhz[r]);
    float nv = tanhf(cnn[r] + rr * dhn[r]);
    Hout[idx] = (1.f - zz) * nv + zz * hv;
  }
}

// ---------------------------------------------------------------------------
// Scatter-add: one wave per edge; lanes stripe the 256 channels (fp32 atomics).
// ---------------------------------------------------------------------------
__global__ __launch_bounds__(256) void scatter_add_kernel(
    const float* __restrict__ mt, const int* __restrict__ esrc,
    const int* __restrict__ edst, const int* __restrict__ etype,
    float* __restrict__ agg, int E, int t) {
  int e = blockIdx.x * (blockDim.x >> 5) + (threadIdx.x >> 5);
  if (e >= E) return;
  if (etype[e] != t) return;
  int lane = threadIdx.x & 31;
  const float* s = mt + (size_t)esrc[e] * DD;
  float* d = agg + (size_t)edst[e] * DD;
#pragma unroll
  for (int c = 0; c < DD / 32; ++c) atomicAdd(&d[lane + 32 * c], s[lane + 32 * c]);
}

// ---------------------------------------------------------------------------
// Small elementwise kernels
// ---------------------------------------------------------------------------
__global__ void copy_kernel(const float* __restrict__ src, float* __restrict__ dst,
                            size_t total) {
  size_t i = (size_t)blockIdx.x * blockDim.x + threadIdx.x;
  if (i < total) dst[i] = src[i];
}

__global__ void zero_kernel(float* __restrict__ dst, size_t total) {
  size_t i = (size_t)blockIdx.x * blockDim.x + threadIdx.x;
  if (i < total) dst[i] = 0.f;
}

// ci_bf[b,n,0:256]=h ; ci_bf[b,n,256:512]=feat   (bf16 output)
__global__ void concat_bf_kernel(const float* __restrict__ H,
                                 const float* __restrict__ F,
                                 __bf16* __restrict__ CI) {
  size_t i = (size_t)blockIdx.x * blockDim.x + threadIdx.x;
  if (i >= (size_t)BN * CCH) return;
  int c = (int)(i & (CCH - 1));
  size_t node = i >> 9;  // CCH == 512
  float v = (c < DD) ? H[node * DD + c] : F[node * DIN + (c - DD)];
  CI[i] = (__bf16)v;
}

// maxpool over L (fp32 in), fp32 out
__global__ void maxpool_f32_kernel(const float* __restrict__ X,
                                   float* __restrict__ Y, int Bn, int Lin, int C,
                                   int win, int stride) {
  int Lout = (Lin - win) / stride + 1;
  size_t total = (size_t)Bn * Lout * C;
  size_t i = (size_t)blockIdx.x * blockDim.x + threadIdx.x;
  if (i >= total) return;
  int c = (int)(i % C);
  size_t t = i / C;
  int l = (int)(t % Lout);
  int b = (int)(t / Lout);
  const float* p = X + ((size_t)b * Lin + (size_t)l * stride) * C + c;
  float m = p[0];
  for (int j = 1; j < win; ++j) { float v = p[(size_t)j * C]; m = v > m ? v : m; }
  Y[i] = m;
}

// maxpool over L (fp32 in), bf16 out (feeds next conv's A operand)
__global__ void maxpool_bf_kernel(const float* __restrict__ X,
                                  __bf16* __restrict__ Y, int Bn, int Lin, int C,
                                  int win, int stride) {
  int Lout = (Lin - win) / stride + 1;
  size_t total = (size_t)Bn * Lout * C;
  size_t i = (size_t)blockIdx.x * blockDim.x + threadIdx.x;
  if (i >= total) return;
  int c = (int)(i % C);
  size_t t = i / C;
  int l = (int)(t % Lout);
  int b = (int)(t / Lout);
  const float* p = X + ((size_t)b * Lin + (size_t)l * stride) * C + c;
  float m = p[0];
  for (int j = 1; j < win; ++j) { float v = p[(size_t)j * C]; m = v > m ? v : m; }
  Y[i] = (__bf16)m;
}

// out[b] = sigmoid( mean_l( (Y2@wy+by) * (Z2@wz+bz) ) ), L = 126
__global__ __launch_bounds__(256) void final_kernel(
    const float* __restrict__ Y2, const float* __restrict__ Z2,
    const float* __restrict__ wy, const float* __restrict__ by,
    const float* __restrict__ wz, const float* __restrict__ bz,
    float* __restrict__ out) {
  __shared__ float sm[256];
  int b = blockIdx.x, tid = threadIdx.x;
  const int L = 126;
  float p = 0.f;
  if (tid < L) {
    const float* yr = Y2 + ((size_t)b * L + tid) * DD;
    float yv = by[0];
    for (int c = 0; c < DD; ++c) yv += yr[c] * wy[c];
    const float* zr = Z2 + ((size_t)b * L + tid) * CCH;
    float zv = bz[0];
    for (int c = 0; c < CCH; ++c) zv += zr[c] * wz[c];
    p = yv * zv;
  }
  sm[tid] = p;
  __syncthreads();
  for (int s = 128; s > 0; s >>= 1) {
    if (tid < s) sm[tid] += sm[tid + s];
    __syncthreads();
  }
  if (tid == 0) out[b] = 1.f / (1.f + __expf(-sm[0] / (float)L));
}

// ---------------------------------------------------------------------------
// Host orchestration
// ---------------------------------------------------------------------------
static inline int tile_blocks(long long tiles) { return (int)((tiles + 7) / 8); }
static inline int elem_blocks(size_t n) { return (int)((n + 255) / 256); }

extern "C" void kernel_launch(void* const* d_in, const int* in_sizes, int n_in,
                              void* d_out, int out_size, void* d_ws, size_t ws_size,
                              hipStream_t stream) {
  const float* feat   = (const float*)d_in[0];
  const int*   esrc   = (const int*)d_in[1];
  const int*   edst   = (const int*)d_in[2];
  const int*   etype  = (const int*)d_in[3];
  const float* W_lin  = (const float*)d_in[4];
  const float* b_lin  = (const float*)d_in[5];
  const float* W_ih   = (const float*)d_in[6];
  const float* W_hh   = (const float*)d_in[7];
  const float* b_ih   = (const float*)d_in[8];
  const float* b_hh   = (const float*)d_in[9];
  const float* conv1_w = (const float*)d_in[10];
  const float* conv1_b = (const float*)d_in[11];
  const float* conv2_w = (const float*)d_in[12];
  const float* conv2_b = (const float*)d_in[13];
  const float* convc1_w = (const float*)d_in[14];
  const float* convc1_b = (const float*)d_in[15];
  const float* convc2_w = (const float*)d_in[16];
  const float* convc2_b = (const float*)d_in[17];
  const float* mlp_y_w = (const float*)d_in[18];
  const float* mlp_y_b = (const float*)d_in[19];
  const float* mlp_z_w = (const float*)d_in[20];
  const float* mlp_z_b = (const float*)d_in[21];
  float* out = (float*)d_out;
  float* ws = (float*)d_ws;
  (void)ws_size; (void)n_in; (void)in_sizes; (void)out_size;

  // ---------------- workspace arena (float offsets) ----------------
  const size_t NH = (size_t)BN * DD;  // 8,355,840
  float* h0  = ws + 0;
  float* h1  = ws + NH;
  float* agg = ws + 2 * NH;
  float* mt  = ws + 3 * NH;
  __bf16* h_bf   = (__bf16*)(ws + 4 * NH);           // NH bf16
  __bf16* agg_bf = (__bf16*)(ws + 4 * NH + NH / 2);  // NH bf16
  __bf16* Wpk    = (__bf16*)(ws + 5 * NH);           // packed weights (~4.5MB)
  // phase 2 (graph buffers dead except h0 / h_bf shadows):
  __bf16* ci_bf     = (__bf16*)(ws + NH);            // 2*NH bf16, over h1
  float*  Y1pre     = ws + 3 * NH;                   // [B,508,256] over mt
  __bf16* Y1pool_bf = (__bf16*)(ws + 2 * NH);        // over agg
  float*  Y2pre     = ws + 2 * NH + 2072576;
  float*  Y2pool    = ws + 2 * NH + 6217728;         // live until final
  float*  Z1pre     = ws + 3 * NH;                   // [B,508,512] over mt/h_bf/agg_bf
  __bf16* Z1pool_bf = (__bf16*)(ws + 0);             // over h0
  float*  Z2pre     = ws + NH;                       // over ci_bf (dead)
  float*  Z2pool    = ws + 4145152;                  // after Z1pool_bf, inside h0 slot

  // packed-weight offsets (bf16 elements)
  __bf16* p_Wlin = Wpk + 0;        // 4 * 65536
  __bf16* p_Wih  = Wpk + 262144;   // 196608
  __bf16* p_Whh  = Wpk + 458752;   // 196608
  __bf16* p_c1   = Wpk + 655360;   // 196608
  __bf16* p_c2   = Wpk + 851968;   // 65536
  __bf16* p_cc1  = Wpk + 917504;   // 786432
  __bf16* p_cc2  = Wpk + 1703936;  // 524288

  // ---------------- pack weights to WMMA B-tile layout ----------------
  for (int t = 0; t < TT; ++t)
    pack_b_kernel<<<16, 256, 0, stream>>>(W_lin + (size_t)t * DD * DD,
                                          p_Wlin + (size_t)t * DD * DD,
                                          16, 8, 1, 256, 1);
  pack_b_kernel<<<48, 256, 0, stream>>>(W_ih, p_Wih, 48, 8, 1, 256, 1);
  pack_b_kernel<<<48, 256, 0, stream>>>(W_hh, p_Whh, 48, 8, 1, 256, 1);
  pack_b_kernel<<<48, 256, 0, stream>>>(conv1_w, p_c1, 16, 8, 3, 768, 3);
  pack_b_kernel<<<16, 256, 0, stream>>>(conv2_w, p_c2, 16, 8, 1, 256, 1);
  pack_b_kernel<<<192, 256, 0, stream>>>(convc1_w, p_cc1, 32, 16, 3, 1536, 3);
  pack_b_kernel<<<128, 256, 0, stream>>>(convc2_w, p_cc2, 32, 16, 2, 1024, 2);

  // h = feat (DIN == D)
  copy_kernel<<<elem_blocks(NH), 256, 0, stream>>>(feat, h0, NH);

  // ---------------- 6 gated message-passing steps ----------------
  const long long mtTiles = (long long)(BN / 16) * (DD / 64);   // 16x64 per wave
  const long long gruTiles = (long long)(BN / 16) * (DD / 16);
  for (int step = 0; step < NSTEPS; ++step) {
    float* hcur = (step & 1) ? h1 : h0;
    float* hnxt = (step & 1) ? h0 : h1;
    pack_a_kernel<<<elem_blocks(NH), 256, 0, stream>>>(hcur, h_bf, NH);
    zero_kernel<<<elem_blocks(NH), 256, 0, stream>>>(agg, NH);
    for (int t = 0; t < TT; ++t) {
      // mt = hcur @ W_lin[t]^T + b_lin[t]  (Ktap==1 GEMM, Bn=1, Lin=BN)
      conv1d_wmma_kernel<<<tile_blocks(mtTiles), 256, 0, stream>>>(
          h_bf, p_Wlin + (size_t)t * DD * DD, b_lin + (size_t)t * DD, mt,
          1, BN, DD, DD, 1, 0);
      scatter_add_kernel<<<(ETOT + 7) / 8, 256, 0, stream>>>(mt, esrc, edst, etype,
                                                             agg, ETOT, t);
    }
    pack_a_kernel<<<elem_blocks(NH), 256, 0, stream>>>(agg, agg_bf, NH);
    gru_wmma_kernel<<<tile_blocks(gruTiles), 256, 0, stream>>>(
        agg_bf, h_bf, hcur, p_Wih, p_Whh, b_ih, b_hh, hnxt, BN);
  }
  // final h in h0

  // bf16 shadows for the conv towers
  pack_a_kernel<<<elem_blocks(NH), 256, 0, stream>>>(h0, h_bf, NH);
  concat_bf_kernel<<<elem_blocks((size_t)BN * CCH), 256, 0, stream>>>(h0, feat, ci_bf);

  // ---------------- Y branch ----------------
  conv1d_wmma_kernel<<<tile_blocks((long long)BB * 32 * 4), 256, 0, stream>>>(
      h_bf, p_c1, conv1_b, Y1pre, BB, NN, DD, DD, 3, 1);            // [B,508,256]
  maxpool_bf_kernel<<<elem_blocks((size_t)BB * 253 * DD), 256, 0, stream>>>(
      Y1pre, Y1pool_bf, BB, 508, DD, 3, 2);
  conv1d_wmma_kernel<<<tile_blocks((long long)BB * 16 * 4), 256, 0, stream>>>(
      Y1pool_bf, p_c2, conv2_b, Y2pre, BB, 253, DD, DD, 1, 1);      // [B,253,256]
  maxpool_f32_kernel<<<elem_blocks((size_t)BB * 126 * DD), 256, 0, stream>>>(
      Y2pre, Y2pool, BB, 253, DD, 2, 2);

  // ---------------- Z branch ----------------
  conv1d_wmma_kernel<<<tile_blocks((long long)BB * 32 * 8), 256, 0, stream>>>(
      ci_bf, p_cc1, convc1_b, Z1pre, BB, NN, CCH, CCH, 3, 1);       // [B,508,512]
  maxpool_bf_kernel<<<elem_blocks((size_t)BB * 253 * CCH), 256, 0, stream>>>(
      Z1pre, Z1pool_bf, BB, 508, CCH, 3, 2);
  conv1d_wmma_kernel<<<tile_blocks((long long)BB * 16 * 8), 256, 0, stream>>>(
      Z1pool_bf, p_cc2, convc2_b, Z2pre, BB, 253, CCH, CCH, 2, 1);  // [B,252,512]
  maxpool_f32_kernel<<<elem_blocks((size_t)BB * 126 * CCH), 256, 0, stream>>>(
      Z2pre, Z2pool, BB, 252, CCH, 2, 2);

  // ---------------- head ----------------
  final_kernel<<<BB, 256, 0, stream>>>(Y2pool, Z2pool, mlp_y_w, mlp_y_b, mlp_z_w,
                                       mlp_z_b, out);
}
